// DecoderVideoRNN_85658827751962
// MI455X (gfx1250) — compile-verified, matrix operand order checked by reading, not run on hardware
//
#include <hip/hip_runtime.h>
#include <hip/hip_bf16.h>

#define B_    64
#define T_ENC 128
#define T_DEC 64
#define DH    1024
#define DV    2048
#define CHUNK 8
#define NCH   256          // DV / CHUNK
#define G     8704         // 4*DV + 2*NCH

typedef __attribute__((ext_vector_type(16))) __bf16 v16bf;
typedef __attribute__((ext_vector_type(8)))  float  v8f;

static __device__ __forceinline__ float sigmoidf_(float x) {
    return 1.0f / (1.0f + __expf(-x));
}

// ---------------------------------------------------------------------------
// One-time conversion kernels
// ---------------------------------------------------------------------------
__global__ void k_cast_bf16(const float* __restrict__ x, __bf16* __restrict__ y, long long n) {
    long long i = (long long)blockIdx.x * blockDim.x + threadIdx.x;
    if (i < n) y[i] = (__bf16)x[i];
}

// W is K x N row-major (fp32); WT is N x K row-major (bf16)
__global__ void k_transpose_bf16(const float* __restrict__ W, __bf16* __restrict__ WT,
                                 int K, int N) {
    long long i = (long long)blockIdx.x * blockDim.x + threadIdx.x;
    long long total = (long long)K * N;
    if (i >= total) return;
    int n = (int)(i / K);
    int k = (int)(i % K);
    WT[i] = (__bf16)W[(long long)k * N + n];
}

__global__ void k_init_state(float* __restrict__ c, __bf16* __restrict__ hB, int n) {
    int i = blockIdx.x * blockDim.x + threadIdx.x;
    if (i < n) { c[i] = 0.0f; hB[i] = (__bf16)0.0f; }
}

// ---------------------------------------------------------------------------
// WMMA GEMM: C[M,N] = A[M,K](bf16,row-major) * B[K,N] (+ bias), where B is
// supplied transposed as BT[N,K] bf16 row-major (B lane order used here, same as A:
// lane l holds N = l%16, K = (l/16)*16 .. +15 contiguous).
// block = 128 threads = 4 waves; each wave -> one 16x16 tile; grid.x covers
// N/64, grid.y covers M/16.
// ---------------------------------------------------------------------------
__global__ void k_wmma_gemm(const __bf16* __restrict__ A, const __bf16* __restrict__ BT,
                            float* __restrict__ C, const float* __restrict__ bias,
                            int N, int K) {
    int lane  = threadIdx.x & 31;
    int wave  = threadIdx.x >> 5;
    int nTile = blockIdx.x * 4 + wave;
    int mTile = blockIdx.y;
    int r16   = lane & 15;
    int kh    = (lane >> 4) << 4;                 // 0 or 16
    const __bf16* ap = A  + (size_t)(mTile * 16 + r16) * K + kh;
    const __bf16* bp = BT + (size_t)(nTile * 16 + r16) * K + kh;
    v8f acc = {};
    for (int k = 0; k < K; k += 32) {
        v16bf a = *(const v16bf*)(ap + k);
        v16bf b = *(const v16bf*)(bp + k);
        acc = __builtin_amdgcn_wmma_f32_16x16x32_bf16(false, a, false, b,
                                                      (short)0, acc, false, false);
    }
    int   col  = nTile * 16 + r16;
    int   row0 = mTile * 16 + ((lane >> 4) << 3); // lanes 16-31 -> rows +8
    float bv   = bias ? bias[col] : 0.0f;
#pragma unroll
    for (int r = 0; r < 8; ++r)
        C[(size_t)(row0 + r) * N + col] = acc[r] + bv;
}

// gates[64,G] = ctx[64,DH]*Wih + h[64,DV]*Whh + (b_ih + b_hh)
__global__ void k_wmma_gates(const __bf16* __restrict__ A1, const __bf16* __restrict__ BT1,
                             const __bf16* __restrict__ A2, const __bf16* __restrict__ BT2,
                             const float* __restrict__ b1, const float* __restrict__ b2,
                             float* __restrict__ C) {
    const int K1 = DH, K2 = DV, N = G;
    int lane  = threadIdx.x & 31;
    int wave  = threadIdx.x >> 5;
    int nTile = blockIdx.x * 4 + wave;
    int mTile = blockIdx.y;
    int r16   = lane & 15;
    int kh    = (lane >> 4) << 4;
    const __bf16* a1 = A1  + (size_t)(mTile * 16 + r16) * K1 + kh;
    const __bf16* p1 = BT1 + (size_t)(nTile * 16 + r16) * K1 + kh;
    v8f acc = {};
    for (int k = 0; k < K1; k += 32) {
        v16bf a = *(const v16bf*)(a1 + k);
        v16bf b = *(const v16bf*)(p1 + k);
        acc = __builtin_amdgcn_wmma_f32_16x16x32_bf16(false, a, false, b,
                                                      (short)0, acc, false, false);
    }
    const __bf16* a2 = A2  + (size_t)(mTile * 16 + r16) * K2 + kh;
    const __bf16* p2 = BT2 + (size_t)(nTile * 16 + r16) * K2 + kh;
    for (int k = 0; k < K2; k += 32) {
        v16bf a = *(const v16bf*)(a2 + k);
        v16bf b = *(const v16bf*)(p2 + k);
        acc = __builtin_amdgcn_wmma_f32_16x16x32_bf16(false, a, false, b,
                                                      (short)0, acc, false, false);
    }
    int   col  = nTile * 16 + r16;
    int   row0 = mTile * 16 + ((lane >> 4) << 3);
    float bv   = b1[col] + b2[col];
#pragma unroll
    for (int r = 0; r < 8; ++r)
        C[(size_t)(row0 + r) * N + col] = acc[r] + bv;
}

// ---------------------------------------------------------------------------
// Block reduction helper (blockDim == 256 == 8 waves of 32)
// ---------------------------------------------------------------------------
__device__ __forceinline__ float blk_reduce(float v, float* red, int tid, bool isMax) {
    int lane = tid & 31, wave = tid >> 5;
#pragma unroll
    for (int o = 16; o; o >>= 1) {
        float t = __shfl_xor(v, o, 32);
        v = isMax ? fmaxf(v, t) : (v + t);
    }
    if (lane == 0) red[wave] = v;
    __syncthreads();
    float r = red[0];
#pragma unroll
    for (int w = 1; w < 8; ++w) r = isMax ? fmaxf(r, red[w]) : (r + red[w]);
    __syncthreads();
    return r;
}

// ---------------------------------------------------------------------------
// Per-step attention: e = tanh(P + hproj) . w2 ; softmax over T_ENC ; context
// One block per batch row, 256 threads.
// ---------------------------------------------------------------------------
__global__ void k_attn(const float* __restrict__ P,    // [B*T_ENC, DH]
                       const float* __restrict__ hp,   // [B, DH]
                       const float* __restrict__ w2,   // [DH]
                       const float* __restrict__ enc,  // [B, T_ENC, DH] fp32
                       __bf16* __restrict__ ctxB) {    // [B, DH] bf16
    __shared__ float e_s[T_ENC];
    __shared__ float red[8];
    int b = blockIdx.x, tid = threadIdx.x;
    int lane = tid & 31, wave = tid >> 5;
    const float* hpb = hp + (size_t)b * DH;
    for (int t = wave; t < T_ENC; t += 8) {
        const float* Pr = P + ((size_t)b * T_ENC + t) * DH;
        float s = 0.0f;
        for (int k = lane; k < DH; k += 32)
            s += tanhf(Pr[k] + hpb[k]) * w2[k];
#pragma unroll
        for (int o = 16; o; o >>= 1) s += __shfl_xor(s, o, 32);
        if (lane == 0) e_s[t] = s;
    }
    __syncthreads();
    float ev  = (tid < T_ENC) ? e_s[tid] : -3.4e38f;
    float m   = blk_reduce(ev, red, tid, true);
    float ex  = (tid < T_ENC) ? __expf(ev - m) : 0.0f;
    float sum = blk_reduce(ex, red, tid, false);
    if (tid < T_ENC) e_s[tid] = ex / sum;
    __syncthreads();
    const float* encb = enc + (size_t)b * T_ENC * DH;
    for (int n = tid; n < DH; n += 256) {
        float a = 0.0f;
#pragma unroll 4
        for (int t = 0; t < T_ENC; ++t)
            a = fmaf(e_s[t], encb[(size_t)t * DH + n], a);
        ctxB[(size_t)b * DH + n] = (__bf16)a;
    }
}

// ---------------------------------------------------------------------------
// ON-LSTM cell: cumsoftmax gates, chunked cell update, write hy to output.
// One block per batch row, 256 threads (== NCH, one thread per chunk).
// ---------------------------------------------------------------------------
__global__ void k_cell(const float* __restrict__ gates,  // [B, G]
                       float* __restrict__ c,            // [B, DV]
                       float* __restrict__ out,          // [B, T_DEC, DV]
                       __bf16* __restrict__ hB,          // [B, DV]
                       int t) {
    __shared__ float s0[NCH];
    __shared__ float s1[NCH];
    __shared__ float red[8];
    int b = blockIdx.x, tid = threadIdx.x;
    const float* gr = gates + (size_t)b * G;

    float x0 = gr[tid];
    float x1 = gr[NCH + tid];
    float m0 = blk_reduce(x0, red, tid, true);
    float e0 = __expf(x0 - m0);
    float z0 = blk_reduce(e0, red, tid, false);
    float p0 = e0 / z0;
    float m1 = blk_reduce(x1, red, tid, true);
    float e1 = __expf(x1 - m1);
    float z1 = blk_reduce(e1, red, tid, false);
    float p1 = e1 / z1;

    s0[tid] = p0; s1[tid] = p1;
    __syncthreads();
    // inclusive scan (Hillis-Steele) over 256 elements, both arrays
    for (int o = 1; o < NCH; o <<= 1) {
        float a0 = (tid >= o) ? s0[tid - o] : 0.0f;
        float a1 = (tid >= o) ? s1[tid - o] : 0.0f;
        __syncthreads();
        s0[tid] += a0; s1[tid] += a1;
        __syncthreads();
    }
    float cin = 1.0f - s0[tid];       // cingate
    float cfg = s1[tid];              // cforgetgate
    float ov  = cfg * cin;
    float fgb = cfg - ov;
    float igb = cin - ov;

    const float* rr = gr + 2 * NCH;   // r = gates[:, 2*NCH:] as [4*NCH, CHUNK]
    size_t  base = (size_t)tid * CHUNK;
    float*  crow = c   + (size_t)b * DV;
    float*  orow = out + ((size_t)b * T_DEC + t) * DV;
    __bf16* hrow = hB  + (size_t)b * DV;
#pragma unroll
    for (int k = 0; k < CHUNK; ++k) {
        float o_ = sigmoidf_(rr[base + k]);                 // o   chunks [0,NCH)
        float g_ = tanhf(rr[(size_t)DV + base + k]);        // g   chunks [NCH,2NCH)
        float i_ = sigmoidf_(rr[2 * (size_t)DV + base + k]);// i
        float f_ = sigmoidf_(rr[3 * (size_t)DV + base + k]);// f
        float fg = f_ * ov + fgb;
        float ig = i_ * ov + igb;
        float cy = fg * crow[base + k] + ig * g_;
        crow[base + k] = cy;
        float hy = o_ * tanhf(cy);
        orow[base + k] = hy;
        hrow[base + k] = (__bf16)hy;
    }
}

// ---------------------------------------------------------------------------
// Host launcher
// ---------------------------------------------------------------------------
extern "C" void kernel_launch(void* const* d_in, const int* in_sizes, int n_in,
                              void* d_out, int out_size, void* d_ws, size_t ws_size,
                              hipStream_t stream) {
    (void)in_sizes; (void)n_in; (void)out_size; (void)ws_size;
    const float* enc = (const float*)d_in[0];   // [B, T_ENC, DH]
    const float* W1  = (const float*)d_in[2];   // [DH+DV, DH]
    const float* b1  = (const float*)d_in[3];   // [DH]
    const float* w2  = (const float*)d_in[4];   // [DH]
    const float* Wih = (const float*)d_in[5];   // [DH, G]
    const float* bih = (const float*)d_in[6];   // [G]
    const float* Whh = (const float*)d_in[7];   // [DV, G]
    const float* bhh = (const float*)d_in[8];   // [G]
    float* out = (float*)d_out;                 // [B, T_DEC, DV]

    size_t off = 0;
    auto alloc = [&](size_t bytes) -> void* {
        void* p = (char*)d_ws + off;
        off += (bytes + 255) & ~(size_t)255;
        return p;
    };
    __bf16* encB  = (__bf16*)alloc((size_t)B_ * T_ENC * DH * 2);  // bf16 encoder
    __bf16* W1eT  = (__bf16*)alloc((size_t)DH * DH * 2);          // [DH][DH]
    __bf16* W1hT  = (__bf16*)alloc((size_t)DH * DV * 2);          // [DH][DV]
    __bf16* WihT  = (__bf16*)alloc((size_t)G * DH * 2);           // [G][DH]
    __bf16* WhhT  = (__bf16*)alloc((size_t)G * DV * 2);           // [G][DV]
    float*  P     = (float*)alloc((size_t)B_ * T_ENC * DH * 4);   // precomputed enc proj
    float*  hp    = (float*)alloc((size_t)B_ * DH * 4);           // h projection
    __bf16* hB    = (__bf16*)alloc((size_t)B_ * DV * 2);          // h state (bf16)
    __bf16* ctxB  = (__bf16*)alloc((size_t)B_ * DH * 2);          // context (bf16)
    float*  gates = (float*)alloc((size_t)B_ * G * 4);
    float*  cst   = (float*)alloc((size_t)B_ * DV * 4);           // cell state

    // ---- one-time conversions ----
    long long nEnc = (long long)B_ * T_ENC * DH;
    k_cast_bf16<<<dim3((unsigned)((nEnc + 255) / 256)), dim3(256), 0, stream>>>(enc, encB, nEnc);
    k_transpose_bf16<<<dim3((unsigned)(((long long)DH * DH + 255) / 256)), dim3(256), 0, stream>>>(
        W1, W1eT, DH, DH);
    k_transpose_bf16<<<dim3((unsigned)(((long long)DV * DH + 255) / 256)), dim3(256), 0, stream>>>(
        W1 + (size_t)DH * DH, W1hT, DV, DH);
    k_transpose_bf16<<<dim3((unsigned)(((long long)DH * G + 255) / 256)), dim3(256), 0, stream>>>(
        Wih, WihT, DH, G);
    k_transpose_bf16<<<dim3((unsigned)(((long long)DV * G + 255) / 256)), dim3(256), 0, stream>>>(
        Whh, WhhT, DV, G);
    k_init_state<<<dim3((B_ * DV + 255) / 256), dim3(256), 0, stream>>>(cst, hB, B_ * DV);

    // ---- step-invariant attention pre-projection:
    // P[b,t,:] = enc[b,t,:] @ W_att1[:DH,:] + b_att1   (M=8192, N=DH, K=DH)
    k_wmma_gemm<<<dim3(DH / 64, (B_ * T_ENC) / 16), dim3(128), 0, stream>>>(
        encB, W1eT, P, b1, DH, DH);

    // ---- 64 recurrent steps ----
    for (int t = 0; t < T_DEC; ++t) {
        // hp = h @ W_att1[DH:,:]   (M=64, N=DH, K=DV)
        k_wmma_gemm<<<dim3(DH / 64, B_ / 16), dim3(128), 0, stream>>>(
            hB, W1hT, hp, nullptr, DH, DV);
        // attention scores + softmax + context
        k_attn<<<dim3(B_), dim3(256), 0, stream>>>(P, hp, w2, enc, ctxB);
        // gates = ctx@Wih + h@Whh + b_ih + b_hh   (M=64, N=G)
        k_wmma_gates<<<dim3(G / 64, B_ / 16), dim3(128), 0, stream>>>(
            ctxB, WihT, hB, WhhT, bih, bhh, gates);
        // ON-LSTM cell update, writes out[:, t, :] and next h (bf16)
        k_cell<<<dim3(B_), dim3(256), 0, stream>>>(gates, cst, out, hB, t);
    }
}
